// MultiHeadAttention_32323923870409
// MI455X (gfx1250) — matrix operand order repaired and checked by reading, not compiled, not run on hardware
//
#include <hip/hip_runtime.h>
#include <hip/hip_bf16.h>

// ---------------------------------------------------------------------------
// MI455X / gfx1250 fused multi-head attention, bf16 WMMA (f32 accumulate).
// ---------------------------------------------------------------------------

typedef __bf16 bf16;
typedef __attribute__((ext_vector_type(16))) __bf16 v16bf;
typedef __attribute__((ext_vector_type(8)))  __bf16 v8bf;
typedef __attribute__((ext_vector_type(4)))  __bf16 v4bf;
typedef __attribute__((ext_vector_type(8)))  float  v8f;

#define D_MODEL 1024
#define NHEAD   16
#define DKH     64
#define BATCH   2
#define SEQ     2048
#define MROWS   (BATCH * SEQ)

static __device__ __forceinline__ v16bf combine8(v8bf lo, v8bf hi) {
  v16bf r;
#pragma unroll
  for (int j = 0; j < 8; ++j) { r[j] = lo[j]; r[j + 8] = hi[j]; }
  return r;
}

// Raw v_exp_f32 (no OCML denorm-range fixup; args are <= 0 in softmax).
static __device__ __forceinline__ float fast_exp2(float x) {
  return __builtin_amdgcn_exp2f(x);
}

// --------------------------- f32 -> bf16 convert ---------------------------
__global__ void cvt_f32_bf16(const float* __restrict__ in,
                             bf16* __restrict__ out, int n4) {
  int i = (blockIdx.x * blockDim.x + threadIdx.x);
  if (i < n4) {
    float4 v = *(const float4*)(in + (size_t)i * 4);
    v4bf o;
    o[0] = (bf16)v.x; o[1] = (bf16)v.y; o[2] = (bf16)v.z; o[3] = (bf16)v.w;
    *(v4bf*)(out + (size_t)i * 4) = o;
  }
}

// --------------------------- WMMA projection GEMM --------------------------
// out[m, n] = (sum_k X[m, k] * W[n, k] + bias[n]) * oscale
// One wave computes a 16(M) x 128(N) tile: A-operand reused across 8 WMMA.
// All loads of a K-step are issued before the WMMA burst.
// MODE 0: store bf16 row-major [MROWS, D]           (q, k projections)
// MODE 1: store bf16 transposed per head [B,H,dk,S] (v projection -> Vt)
// MODE 2: store f32 row-major [MROWS, D]            (final output projection)
template <int MODE>
__global__ __launch_bounds__(32) void proj16x128(const bf16* __restrict__ X,
                                                 const bf16* __restrict__ W,
                                                 const float* __restrict__ bias,
                                                 bf16* __restrict__ outb,
                                                 float* __restrict__ outf,
                                                 float oscale) {
  const int lane = threadIdx.x & 31;
  const int col  = lane & 15;
  const int grp  = lane >> 4;
  const int mt   = blockIdx.x * 16;
  const int nb   = blockIdx.y * 128;

  v8f acc[8] = {};
  const bf16* xrow = X + (size_t)(mt + col) * D_MODEL;

#pragma unroll 2
  for (int k0 = 0; k0 < D_MODEL; k0 += 32) {
    // A-layout (16-bit, 16x32): lane = row m, K = 16*grp + j  (contiguous 32B)
    v16bf a = *(const v16bf*)(xrow + k0 + grp * 16);
    // B-layout (32x16): lane = column n, K = 16*grp + j  (contiguous 32B)
    v16bf bm[8];
#pragma unroll
    for (int nt = 0; nt < 8; ++nt)
      bm[nt] = *(const v16bf*)(W + (size_t)(nb + nt * 16 + col) * D_MODEL +
                               k0 + grp * 16);
#pragma unroll
    for (int nt = 0; nt < 8; ++nt)
      acc[nt] = __builtin_amdgcn_wmma_f32_16x16x32_bf16(
          false, a, false, bm[nt], (short)0, acc[nt], false, false);
  }

#pragma unroll
  for (int nt = 0; nt < 8; ++nt) {
    const int   n  = nb + nt * 16 + col;
    const float bv = bias[n];
#pragma unroll
    for (int r = 0; r < 8; ++r) {
      // C-layout: row m = mt + r + 8*grp, col n = lane&15
      const int   m   = mt + r + 8 * grp;
      const float val = (acc[nt][r] + bv) * oscale;
      if (MODE == 0) {
        outb[(size_t)m * D_MODEL + n] = (bf16)val;
      } else if (MODE == 1) {
        const int bb = m >> 11, s = m & (SEQ - 1);
        const int h = n >> 6, dk = n & (DKH - 1);
        outb[(((size_t)(bb * NHEAD + h)) * DKH + dk) * SEQ + s] = (bf16)val;
      } else {
        outf[(size_t)m * D_MODEL + n] = val;
      }
    }
  }
}

// ------------------------ fused streaming attention ------------------------
// One wave handles one (b, h, 16-query tile). Streams keys 32 at a time:
//   all 16 b128 loads issued first (V loads hide under softmax VALU)
//   S^T(32k x 16q) = K_tile x Q^T   (4 WMMA, dk contraction in 2 chunks)
//   online softmax; q was pre-scaled by log2(e)/sqrt(dk), so scores are
//   already in the exp2 domain: max path is mul-free, exp = v_sub + v_exp_f32
//   ctx^T(64dk x 16q) += V^T x P    (4 WMMA, P re-laid out with shfl_xor(16))
__global__ __launch_bounds__(32) void attn_fused(const bf16* __restrict__ qb,
                                                 const bf16* __restrict__ kb,
                                                 const bf16* __restrict__ vt,
                                                 bf16* __restrict__ ctx) {
  const int lane = threadIdx.x & 31;
  const int col  = lane & 15;
  const int grp  = lane >> 4;
  const int qt   = blockIdx.x * 16;
  const int h    = blockIdx.y;
  const int b    = blockIdx.z;

  const size_t rowQ  = ((size_t)(b * SEQ + qt + col)) * D_MODEL + h * DKH;
  const size_t baseK = ((size_t)(b * SEQ)) * D_MODEL + h * DKH;
  const size_t baseV = ((size_t)(b * NHEAD + h)) * DKH;  // row idx into [.,SEQ]

  // B-operand Q^T, two dk-chunks (dk 0..31 / 32..63): 32B contiguous per lane
  v16bf bq0 = *(const v16bf*)(qb + rowQ + grp * 16);
  v16bf bq1 = *(const v16bf*)(qb + rowQ + 32 + grp * 16);

  v8f   O[4]  = {};
  float m_run = -1e30f, l_run = 0.f;

  for (int k0 = 0; k0 < SEQ; k0 += 32) {
    if (k0 + 32 < SEQ) {  // uniform branch: EXEC stays all-ones
      __builtin_prefetch(kb + baseK + (size_t)(k0 + 32 + col) * D_MODEL, 0, 1);
      __builtin_prefetch(vt + (baseV + col) * (size_t)SEQ + k0 + 32, 0, 1);
    }

    // ---- issue ALL global loads of this step up front ----
    v16bf aK[2][2];  // [key-tile t][dk-chunk c]
#pragma unroll
    for (int t = 0; t < 2; ++t) {
      const bf16* krow = kb + baseK + (size_t)(k0 + t * 16 + col) * D_MODEL;
#pragma unroll
      for (int c = 0; c < 2; ++c)
        aK[t][c] = *(const v16bf*)(krow + c * 32 + grp * 16);
    }
    v16bf aV[4];     // [dk-chunk of 16]
#pragma unroll
    for (int c4 = 0; c4 < 4; ++c4) {
      const bf16* vrow =
          vt + (baseV + c4 * 16 + col) * (size_t)SEQ + k0 + grp * 16;
      aV[c4] = *(const v16bf*)(vrow);
    }

    // ---- scores: two 16-key tiles, S^T = K x Q^T (exp2-domain already) ----
    v8f s[2] = {};
#pragma unroll
    for (int t = 0; t < 2; ++t) {
      s[t] = __builtin_amdgcn_wmma_f32_16x16x32_bf16(
          false, aK[t][0], false, bq0, (short)0, s[t], false, false);
      s[t] = __builtin_amdgcn_wmma_f32_16x16x32_bf16(
          false, aK[t][1], false, bq1, (short)0, s[t], false, false);
    }

    // ---- online softmax over the 32 new keys (per-q = per-lane stats) ----
    float mloc = -1e30f;
#pragma unroll
    for (int t = 0; t < 2; ++t)
#pragma unroll
      for (int r = 0; r < 8; ++r) mloc = fmaxf(mloc, s[t][r]);
    mloc = fmaxf(mloc, __shfl_xor(mloc, 16, 32));
    const float m_new = fmaxf(m_run, mloc);
    const float corr  = fast_exp2(m_run - m_new);

    v8f p[2];
    float rsum = 0.f;
#pragma unroll
    for (int t = 0; t < 2; ++t)
#pragma unroll
      for (int r = 0; r < 8; ++r) {
        p[t][r] = fast_exp2(s[t][r] - m_new);
        rsum += p[t][r];
      }
    rsum += __shfl_xor(rsum, 16, 32);
    l_run = l_run * corr + rsum;
    m_run = m_new;
#pragma unroll
    for (int c4 = 0; c4 < 4; ++c4)
#pragma unroll
      for (int r = 0; r < 8; ++r) O[c4][r] *= corr;

    // ---- repack P (S^T C-layout) into B-layout: K = 16*grp + j ----
    v16bf bp;
#pragma unroll
    for (int j = 0; j < 8; ++j) {
      const float o0 = __shfl_xor(p[0][j], 16, 32);
      const float o1 = __shfl_xor(p[1][j], 16, 32);
      bp[j]     = (bf16)(grp ? o1 : p[0][j]);
      bp[j + 8] = (bf16)(grp ? p[1][j] : o0);
    }

    // ---- context: ctx^T += V^T x P, 4 dk-chunks of 16 ----
#pragma unroll
    for (int c4 = 0; c4 < 4; ++c4)
      O[c4] = __builtin_amdgcn_wmma_f32_16x16x32_bf16(
          false, aV[c4], false, bp, (short)0, O[c4], false, false);
  }

  // ---- normalize and repack ctx^T -> ctx[b, s, h*64+dk] via LDS ----
  const float inv = 1.f / l_run;
  __shared__ float lds[16 * 65];  // pad 64->65 to dodge bank conflicts
#pragma unroll
  for (int c4 = 0; c4 < 4; ++c4)
#pragma unroll
    for (int r = 0; r < 8; ++r)
      lds[col * 65 + c4 * 16 + r + 8 * grp] = O[c4][r] * inv;
  __syncthreads();

#pragma unroll
  for (int it = 0; it < 8; ++it) {
    const int flat = it * 128 + lane * 4;
    const int row = flat >> 6, d0 = flat & 63;
    v4bf o;
    o[0] = (bf16)lds[row * 65 + d0 + 0];
    o[1] = (bf16)lds[row * 65 + d0 + 1];
    o[2] = (bf16)lds[row * 65 + d0 + 2];
    o[3] = (bf16)lds[row * 65 + d0 + 3];
    *(v4bf*)(ctx + ((size_t)(b * SEQ + qt + row)) * D_MODEL + h * DKH + d0) = o;
  }
}

// --------------------------------- launch ----------------------------------
extern "C" void kernel_launch(void* const* d_in, const int* in_sizes, int n_in,
                              void* d_out, int out_size, void* d_ws,
                              size_t ws_size, hipStream_t stream) {
  (void)in_sizes; (void)n_in; (void)out_size; (void)ws_size;
  const float* Q  = (const float*)d_in[0];
  const float* K  = (const float*)d_in[1];
  const float* V  = (const float*)d_in[2];
  const float* Wq = (const float*)d_in[3];
  const float* bq = (const float*)d_in[4];
  const float* Wk = (const float*)d_in[5];
  const float* bk = (const float*)d_in[6];
  const float* Wv = (const float*)d_in[7];
  const float* bv = (const float*)d_in[8];
  const float* Wo = (const float*)d_in[9];
  const float* bo = (const float*)d_in[10];
  float* out = (float*)d_out;

  const size_t NX = (size_t)MROWS * D_MODEL;   // 4M elems
  const size_t NW = (size_t)D_MODEL * D_MODEL; // 1M elems

  char*  ws  = (char*)d_ws;
  size_t off = 0;
  auto bump = [&](size_t bytes) {
    void* p = ws + off;
    off += (bytes + 255) & ~(size_t)255;
    return p;
  };
  bf16* Qb   = (bf16*)bump(NX * 2);
  bf16* Kb   = (bf16*)bump(NX * 2);
  bf16* Vb   = (bf16*)bump(NX * 2);
  bf16* Wqb  = (bf16*)bump(NW * 2);
  bf16* Wkb  = (bf16*)bump(NW * 2);
  bf16* Wvb  = (bf16*)bump(NW * 2);
  bf16* Wob  = (bf16*)bump(NW * 2);
  bf16* qbf  = (bf16*)bump(NX * 2);  // [MROWS, D]  (pre-scaled, exp2 domain)
  bf16* kbf  = (bf16*)bump(NX * 2);  // [MROWS, D]
  bf16* vtb  = (bf16*)bump(NX * 2);  // [B, H, dk, S]
  bf16* ctxb = (bf16*)bump(NX * 2);  // [MROWS, D]

  // 1) one-shot precision conversion (pure-bf16 GEMM loops afterwards)
  const int tX = (int)(NX / 4), tW = (int)(NW / 4);
  cvt_f32_bf16<<<(tX + 255) / 256, 256, 0, stream>>>(Q, Qb, tX);
  cvt_f32_bf16<<<(tX + 255) / 256, 256, 0, stream>>>(K, Kb, tX);
  cvt_f32_bf16<<<(tX + 255) / 256, 256, 0, stream>>>(V, Vb, tX);
  cvt_f32_bf16<<<(tW + 255) / 256, 256, 0, stream>>>(Wq, Wqb, tW);
  cvt_f32_bf16<<<(tW + 255) / 256, 256, 0, stream>>>(Wk, Wkb, tW);
  cvt_f32_bf16<<<(tW + 255) / 256, 256, 0, stream>>>(Wv, Wvb, tW);
  cvt_f32_bf16<<<(tW + 255) / 256, 256, 0, stream>>>(Wo, Wob, tW);

  // 2) projections (one wave per 16x128 output tile)
  //    q gets 1/sqrt(dk)*log2(e) folded in -> scores land in exp2 domain
  const float qscale = 0.125f * 1.44269504088896341f;
  dim3 gridP(MROWS / 16, D_MODEL / 128);
  proj16x128<0><<<gridP, 32, 0, stream>>>(Qb, Wqb, bq, qbf, nullptr, qscale);
  proj16x128<0><<<gridP, 32, 0, stream>>>(Kb, Wkb, bk, kbf, nullptr, 1.0f);
  proj16x128<1><<<gridP, 32, 0, stream>>>(Vb, Wvb, bv, vtb, nullptr, 1.0f);

  // 3) fused streaming attention (one wave per 16-query tile per head)
  dim3 gridA(SEQ / 16, NHEAD, BATCH);
  attn_fused<<<gridA, 32, 0, stream>>>(qbf, kbf, vtb, ctxb);

  // 4) output projection, f32 result + bias
  proj16x128<2><<<gridP, 32, 0, stream>>>(ctxb, Wob, bo, nullptr, out, 1.0f);
}
